// GATJumpMGEmbedderAR_15178414424417
// MI455X (gfx1250) — compile-verified, matrix operand order checked by reading, not run on hardware
//
#include <hip/hip_runtime.h>
#include <hip/hip_bf16.h>
#include <math.h>

#define NNODES 16384
#define NGRAPH 16
#define NEDGE  131072
#define NHEAD  2
#define NPG    (NNODES / NGRAPH)   // 1024
#define OUTDIM 896                 // 128+256+512

typedef __attribute__((ext_vector_type(16))) __bf16 v16bf;
typedef __attribute__((ext_vector_type(8)))  float  v8f;
typedef __attribute__((ext_vector_type(4)))  unsigned tdm_g0_t;
typedef __attribute__((ext_vector_type(8)))  int      tdm_g1_t;
typedef __attribute__((ext_vector_type(4)))  int      tdm_g2_t;

// ---------- helpers ----------
__device__ __forceinline__ __bf16 f2bf(float f) {
  unsigned u = __float_as_uint(f);
  unsigned r = u + 0x7FFFu + ((u >> 16) & 1u);   // round-to-nearest-even
  unsigned short h = (unsigned short)(r >> 16);
  __bf16 b;
  __builtin_memcpy(&b, &h, sizeof(b));
  return b;
}
__device__ __forceinline__ unsigned ordf(float f) {   // order-preserving uint
  unsigned u = __float_as_uint(f);
  return (u & 0x80000000u) ? ~u : (u | 0x80000000u);
}
__device__ __forceinline__ float unordf(unsigned v) {
  unsigned u = (v & 0x80000000u) ? (v & 0x7FFFFFFFu) : ~v;
  return __uint_as_float(u);
}
__device__ __forceinline__ float lrelu(float x, float s) { return x > 0.f ? x : s * x; }

typedef __attribute__((address_space(3))) char lds_char_t;
__device__ __forceinline__ unsigned lds_offset(void* p) {
  // generic -> LDS addrspace, ptrtoint yields byte offset from workgroup LDS base
  return (unsigned)(size_t)(lds_char_t*)p;
}

// ---------- TDM: load a 2-D tile (tile_h rows x tile_w bf16) into LDS ----------
// Row r of the tile comes from gptr + r*row_stride_elems*2 bytes.
// LDS padding: +4 DWORDs after every 16 DWORDs stored (row = 32 bf16 = 16 DW),
// reproducing an LDS row stride of 40 bf16.
__device__ __forceinline__ void tdm_load_tile(
    unsigned lds_addr, const __bf16* gptr, unsigned row_stride_elems,
    unsigned tile_w, unsigned tile_h)
{
  unsigned long long ga = (unsigned long long)(size_t)gptr;
  tdm_g0_t g0;
  g0[0] = 1u;                                            // count=1, no gather
  g0[1] = lds_addr;                                      // lds_addr
  g0[2] = (unsigned)(ga & 0xFFFFFFFFu);                  // global_addr[31:0]
  g0[3] = (unsigned)((ga >> 32) & 0x01FFFFFFu)           // global_addr[56:32]
        | (2u << 30);                                    // type = 2 ("image")

  const unsigned td0 = row_stride_elems;                 // tensor_dim0 (no OOB hit)
  const unsigned td1 = 1u << 20;                         // tensor_dim1 (large)
  tdm_g1_t g1;
  g1[0] = (int)((1u << 16)        // data_size = 2 bytes (bf16)
              | (1u << 20)        // pad_enable
              | (3u << 22)        // pad_interval: 16 DWORDs
              | (3u << 25));      // pad_amount:   4 DWORDs
  g1[1] = (int)((td0 & 0xFFFFu) << 16);                             // dim0[15:0]
  g1[2] = (int)(((td0 >> 16) & 0xFFFFu) | ((td1 & 0xFFFFu) << 16)); // dim0 hi, dim1 lo
  g1[3] = (int)(((td1 >> 16) & 0xFFFFu) | ((tile_w & 0xFFFFu) << 16)); // dim1 hi, tile_dim0
  g1[4] = (int)(tile_h & 0xFFFFu);                                  // tile_dim1 (tile_dim2=0)
  g1[5] = (int)row_stride_elems;                                    // dim0_stride[31:0]
  g1[6] = 0;                                                        // dim0_stride hi, dim1_stride lo
  g1[7] = 0;
  tdm_g2_t gz = {};                                      // 2-D tensor: groups 2/3 unused
#if __clang_major__ >= 23
  tdm_g1_t z8 = {};
  __builtin_amdgcn_tensor_load_to_lds(g0, g1, gz, gz, z8, 0);
#else
  __builtin_amdgcn_tensor_load_to_lds(g0, g1, gz, gz, 0);
#endif
}

// ---------- bf16 WMMA GEMM, double-buffered TDM staging ----------
// C[M,Ncol] = A[M,K] * Bt[Ncol,K]^T   (A, Bt bf16 row-major; C f32)
#define BK 32
#define LPAD 8
#define LSTR (BK + LPAD)   // 40 bf16 per LDS row (80B) == TDM pad pattern
#define TILE_ELEMS (64 * LSTR + LPAD)

__global__ __launch_bounds__(256) void wmma_gemm_tdm(
    const __bf16* __restrict__ A, const __bf16* __restrict__ Bt,
    float* __restrict__ C, int M, int K, int Ncol)
{
  __shared__ __align__(16) __bf16 sA[2][TILE_ELEMS];   // [buf][m][k]
  __shared__ __align__(16) __bf16 sB[2][TILE_ELEMS];   // [buf][n][k]

  const int tid  = threadIdx.x;
  const int lane = tid & 31;
  const int wave = tid >> 5;
  const int wm   = wave & 3;    // M band (0..3) * 16
  const int wn   = wave >> 2;   // N band (0..1) * 32

  const int row0 = blockIdx.y * 64;
  const int col0 = blockIdx.x * 64;

  v8f acc0 = {}; v8f acc1 = {};

  const int mr = lane & 15;
  const int kb = (lane >> 4) << 3;   // 0 or 8 (ISA 16-bit fragment layout)

  const unsigned ldsA0 = lds_offset(&sA[0][0]);
  const unsigned ldsA1 = lds_offset(&sA[1][0]);
  const unsigned ldsB0 = lds_offset(&sB[0][0]);
  const unsigned ldsB1 = lds_offset(&sB[1][0]);

  const __bf16* Ab = A  + (size_t)row0 * K;
  const __bf16* Bb = Bt + (size_t)col0 * K;

  if (wave == 0) {   // preload tile 0 into buffer 0
    tdm_load_tile(ldsA0, Ab, (unsigned)K, BK, 64);
    tdm_load_tile(ldsB0, Bb, (unsigned)K, BK, 64);
  }

  int cur = 0;
  for (int k0 = 0; k0 < K; k0 += BK) {
    if (wave == 0) {
      int kn = k0 + BK;
      if (kn < K) {          // prefetch next tile into the other buffer
        tdm_load_tile(cur ? ldsA0 : ldsA1, Ab + kn, (unsigned)K, BK, 64);
        tdm_load_tile(cur ? ldsB0 : ldsB1, Bb + kn, (unsigned)K, BK, 64);
        // in-order completion: <=2 outstanding leaves only the 2 new ops in flight
        __builtin_amdgcn_s_wait_tensorcnt((short)2);
      } else {
        __builtin_amdgcn_s_wait_tensorcnt((short)0);
      }
    }
    __syncthreads();   // current buffer visible to all waves

    union { v16bf v; uint4 q[2]; } af, bf0, bf1;
    const __bf16* pa = &sA[cur][(wm * 16 + mr) * LSTR + kb];
    af.q[0] = *(const uint4*)(pa);
    af.q[1] = *(const uint4*)(pa + 16);
    const __bf16* pb0 = &sB[cur][(wn * 32 + mr) * LSTR + kb];
    bf0.q[0] = *(const uint4*)(pb0);
    bf0.q[1] = *(const uint4*)(pb0 + 16);
    const __bf16* pb1 = pb0 + 16 * LSTR;
    bf1.q[0] = *(const uint4*)(pb1);
    bf1.q[1] = *(const uint4*)(pb1 + 16);

    acc0 = __builtin_amdgcn_wmma_f32_16x16x32_bf16(false, af.v, false, bf0.v,
                                                   (short)0, acc0, false, false);
    acc1 = __builtin_amdgcn_wmma_f32_16x16x32_bf16(false, af.v, false, bf1.v,
                                                   (short)0, acc1, false, false);
    __syncthreads();   // all reads of 'cur' done before it is re-targeted by TDM
    cur ^= 1;
  }

  // C/D layout: VGPR i -> M = i (+8 for lanes 16..31), N = lane&15
  const int rbase = row0 + wm * 16 + ((lane >> 4) << 3);
  const int cbase = col0 + wn * 32 + (lane & 15);
#pragma unroll
  for (int i = 0; i < 8; ++i) {
    C[(size_t)(rbase + i) * Ncol + cbase]        = acc0[i];
    C[(size_t)(rbase + i) * Ncol + (cbase + 16)] = acc1[i];
  }
}

// ---------- precision conversion ----------
__global__ void cvt_f32_to_bf16(const float* __restrict__ in, __bf16* __restrict__ out, int n) {
  int t = blockIdx.x * blockDim.x + threadIdx.x;
  if (t < n) out[t] = f2bf(in[t]);
}
// Wt[n*K + k] = bf16(W[k*HD + n])  (coalesced reads)
__global__ void cvt_w_transpose(const float* __restrict__ W, __bf16* __restrict__ Wt,
                                int K, int HD) {
  int t = blockIdx.x * blockDim.x + threadIdx.x;
  if (t >= K * HD) return;
  int n = t % HD, k = t / HD;
  Wt[(size_t)n * K + k] = f2bf(W[t]);
}

// ---------- attention logits: el/er[n,h] = sum_c h[n,h,c] * a{l,r}[h,c] ----------
__global__ __launch_bounds__(256) void attn_logits(
    const float* __restrict__ h, const float* __restrict__ al, const float* __restrict__ ar,
    float* __restrict__ el, float* __restrict__ er, int d)
{
  int lane = threadIdx.x & 31;
  int wid  = (int)((blockIdx.x * blockDim.x + threadIdx.x) >> 5);
  if (wid >= NNODES * NHEAD) return;
  int hh = wid & (NHEAD - 1);
  int n  = wid >> 1;
  const float* hp  = h + (size_t)n * NHEAD * d + hh * d;
  const float* alp = al + hh * d;
  const float* arp = ar + hh * d;
  float sl = 0.f, sr = 0.f;
  for (int c = lane; c < d; c += 32) { float v = hp[c]; sl += v * alp[c]; sr += v * arp[c]; }
#pragma unroll
  for (int off = 16; off; off >>= 1) {
    sl += __shfl_down(sl, off, 32);
    sr += __shfl_down(sr, off, 32);
  }
  if (lane == 0) { el[wid] = sl; er[wid] = sr; }
}

// ---------- edge softmax ----------
__global__ void edge_max(const int* __restrict__ src, const int* __restrict__ dst,
                         const float* __restrict__ el, const float* __restrict__ er,
                         unsigned* __restrict__ mbuf)
{
  int t = blockIdx.x * blockDim.x + threadIdx.x;
  if (t >= NEDGE * NHEAD) return;
  int hh = t & 1, e = t >> 1;
  int s = src[e], dn = dst[e];
  float v = lrelu(el[s * NHEAD + hh] + er[dn * NHEAD + hh], 0.2f);
  atomicMax(&mbuf[dn * NHEAD + hh], ordf(v));
}

__global__ void edge_expsum(const int* __restrict__ src, const int* __restrict__ dst,
                            const float* __restrict__ el, const float* __restrict__ er,
                            const unsigned* __restrict__ mbuf, float* __restrict__ sbuf)
{
  int t = blockIdx.x * blockDim.x + threadIdx.x;
  if (t >= NEDGE * NHEAD) return;
  int hh = t & 1, e = t >> 1;
  int s = src[e], dn = dst[e];
  int idx = dn * NHEAD + hh;
  float v = lrelu(el[s * NHEAD + hh] + er[idx], 0.2f);
  atomicAdd(&sbuf[idx], expf(v - unordf(mbuf[idx])));
}

__global__ __launch_bounds__(256) void edge_aggregate(
    const int* __restrict__ src, const int* __restrict__ dst,
    const float* __restrict__ el, const float* __restrict__ er,
    const unsigned* __restrict__ mbuf, const float* __restrict__ sbuf,
    const float* __restrict__ hfeat, float* __restrict__ out, int d)
{
  int lane = threadIdx.x & 31;
  int wid  = (int)((blockIdx.x * blockDim.x + threadIdx.x) >> 5);
  if (wid >= NEDGE * NHEAD) return;
  int hh = wid & 1, e = wid >> 1;
  int s = src[e], dn = dst[e];
  int idx = dn * NHEAD + hh;
  float v = lrelu(el[s * NHEAD + hh] + er[idx], 0.2f);
  float a = expf(v - unordf(mbuf[idx])) / sbuf[idx];
  const float* hp = hfeat + (size_t)s * NHEAD * d + hh * d;
  float* op       = out   + (size_t)dn * NHEAD * d + hh * d;
  for (int c = lane; c < d; c += 32)
    atomicAdd(&op[c], hp[c] * a);
}

// ---------- GraphNorm stats ----------
__global__ __launch_bounds__(256) void chan_reduce_sum(
    const float* __restrict__ x, float* __restrict__ out, int d)
{
  __shared__ float s[512];
  for (int c = threadIdx.x; c < d; c += 256) s[c] = 0.f;
  __syncthreads();
  size_t base = (size_t)blockIdx.x * 32 * NHEAD * d;
  int total = 32 * NHEAD * d;
  for (int i = threadIdx.x; i < total; i += 256)
    atomicAdd(&s[i % d], x[base + i]);
  __syncthreads();
  for (int c = threadIdx.x; c < d; c += 256) atomicAdd(&out[c], s[c]);
}

__global__ __launch_bounds__(256) void chan_reduce_var(
    const float* __restrict__ x, const float* __restrict__ mean,
    const float* __restrict__ alpha, float* __restrict__ out, int d)
{
  __shared__ float s[512];
  for (int c = threadIdx.x; c < d; c += 256) s[c] = 0.f;
  __syncthreads();
  size_t base = (size_t)blockIdx.x * 32 * NHEAD * d;
  int total = 32 * NHEAD * d;
  for (int i = threadIdx.x; i < total; i += 256) {
    int c = i % d;
    float sh = x[base + i] - alpha[c] * mean[c];
    atomicAdd(&s[c], sh * sh);
  }
  __syncthreads();
  for (int c = threadIdx.x; c < d; c += 256) atomicAdd(&out[c], s[c]);
}

__global__ void fin_mean(const float* __restrict__ csum, float* __restrict__ mean, int d) {
  int c = blockIdx.x * blockDim.x + threadIdx.x;
  if (c < d) mean[c] = csum[c] * (1.0f / (NNODES * NHEAD));
}
__global__ void fin_rstd(const float* __restrict__ cvar, float* __restrict__ rstd, int d) {
  int c = blockIdx.x * blockDim.x + threadIdx.x;
  if (c < d) rstd[c] = rsqrtf(cvar[c] * (1.0f / (NNODES * NHEAD)) + 1e-5f);
}

// ---------- normalize + leaky + head-mean + per-graph pool (writes bf16 x_next) ----------
__global__ __launch_bounds__(256) void apply_norm_pool(
    const float* __restrict__ xin, const float* __restrict__ mean, const float* __restrict__ rstd,
    const float* __restrict__ gamma, const float* __restrict__ beta, const float* __restrict__ alpha,
    __bf16* __restrict__ xout, float* __restrict__ pool, int d, int pool_off)
{
  int t = blockIdx.x * blockDim.x + threadIdx.x;
  if (t >= NNODES * d) return;
  int c = t % d, n = t / d;
  float mu = alpha[c] * mean[c];
  float rs = rstd[c], g = gamma[c], b = beta[c];
  size_t base = (size_t)n * NHEAD * d;
  float y0 = lrelu(g * (xin[base + c]     - mu) * rs + b, 0.01f);
  float y1 = lrelu(g * (xin[base + d + c] - mu) * rs + b, 0.01f);
  xout[base + c]     = f2bf(y0);     // next layer consumes bf16 directly
  xout[base + d + c] = f2bf(y1);
  float nm = 0.5f * (y0 + y1);
  atomicAdd(&pool[(n / NPG) * OUTDIM + pool_off + c], nm * (1.0f / NPG));
}

__global__ void final_out(const float* __restrict__ pool, float* __restrict__ out) {
  int t = blockIdx.x * blockDim.x + threadIdx.x;
  if (t < NGRAPH * OUTDIM) out[t] = lrelu(pool[t], 0.01f);
}

// ---------- host orchestration ----------
extern "C" void kernel_launch(void* const* d_in, const int* in_sizes, int n_in,
                              void* d_out, int out_size, void* d_ws, size_t ws_size,
                              hipStream_t stream)
{
  (void)in_sizes; (void)n_in; (void)out_size; (void)ws_size;
  const float* node_feats = (const float*)d_in[0];
  const int*   esrc = (const int*)d_in[22];
  const int*   edst = (const int*)d_in[23];

  char* ws = (char*)d_ws;
  size_t off = 0;
  auto alloc = [&](size_t bytes) -> void* {
    void* p = ws + off;
    off += (bytes + 255) & ~(size_t)255;
    return p;
  };
  __bf16*   Xbf  = (__bf16*)alloc((size_t)NNODES * 1024 * 2);   // current features (bf16)
  __bf16*   Wtb  = (__bf16*)alloc((size_t)1024 * 512 * 2);      // W^T  bf16
  __bf16*   rWtb = (__bf16*)alloc((size_t)1024 * 512 * 2);      // rW^T bf16
  float*    Hbuf = (float*)alloc((size_t)NNODES * 1024 * 4);
  float*    Obuf = (float*)alloc((size_t)NNODES * 1024 * 4);
  float*    elb  = (float*)alloc((size_t)NNODES * NHEAD * 4);
  float*    erb  = (float*)alloc((size_t)NNODES * NHEAD * 4);
  unsigned* mbuf = (unsigned*)alloc((size_t)NNODES * NHEAD * 4);
  float*    sbuf = (float*)alloc((size_t)NNODES * NHEAD * 4);
  float*    cs   = (float*)alloc(512 * 4);
  float*    cv   = (float*)alloc(512 * 4);
  float*    mnb  = (float*)alloc(512 * 4);
  float*    rsb  = (float*)alloc(512 * 4);
  float*    pool = (float*)alloc((size_t)NGRAPH * OUTDIM * 4);

  cvt_f32_to_bf16<<<(NNODES * 128 + 255) / 256, 256, 0, stream>>>(node_feats, Xbf,
                                                                  NNODES * 128);
  hipMemsetAsync(pool, 0, (size_t)NGRAPH * OUTDIM * 4, stream);

  const int dims[3] = {128, 256, 512};
  int pool_off = 0;
  for (int l = 0; l < 3; ++l) {
    const int d   = dims[l];
    const int HD  = NHEAD * d;     // output width; input width == d
    const float* W  = (const float*)d_in[1 + 7 * l + 0];
    const float* al = (const float*)d_in[1 + 7 * l + 1];
    const float* ar = (const float*)d_in[1 + 7 * l + 2];
    const float* rW = (const float*)d_in[1 + 7 * l + 3];
    const float* gm = (const float*)d_in[1 + 7 * l + 4];
    const float* bt = (const float*)d_in[1 + 7 * l + 5];
    const float* ap = (const float*)d_in[1 + 7 * l + 6];

    cvt_w_transpose<<<(d * HD + 255) / 256, 256, 0, stream>>>(W,  Wtb,  d, HD);
    cvt_w_transpose<<<(d * HD + 255) / 256, 256, 0, stream>>>(rW, rWtb, d, HD);

    dim3 gg(HD / 64, NNODES / 64);
    wmma_gemm_tdm<<<gg, 256, 0, stream>>>(Xbf, Wtb,  Hbuf, NNODES, d, HD);
    wmma_gemm_tdm<<<gg, 256, 0, stream>>>(Xbf, rWtb, Obuf, NNODES, d, HD); // residual init

    attn_logits<<<(NNODES * NHEAD) / 8, 256, 0, stream>>>(Hbuf, al, ar, elb, erb, d);

    hipMemsetAsync(mbuf, 0, (size_t)NNODES * NHEAD * 4, stream); // ord(-inf-ish)
    hipMemsetAsync(sbuf, 0, (size_t)NNODES * NHEAD * 4, stream);
    edge_max   <<<(NEDGE * NHEAD + 255) / 256, 256, 0, stream>>>(esrc, edst, elb, erb, mbuf);
    edge_expsum<<<(NEDGE * NHEAD + 255) / 256, 256, 0, stream>>>(esrc, edst, elb, erb, mbuf, sbuf);
    edge_aggregate<<<(NEDGE * NHEAD) / 8, 256, 0, stream>>>(esrc, edst, elb, erb, mbuf, sbuf,
                                                            Hbuf, Obuf, d);

    hipMemsetAsync(cs, 0, d * 4, stream);
    chan_reduce_sum<<<NNODES / 32, 256, 0, stream>>>(Obuf, cs, d);
    fin_mean<<<(d + 255) / 256, 256, 0, stream>>>(cs, mnb, d);
    hipMemsetAsync(cv, 0, d * 4, stream);
    chan_reduce_var<<<NNODES / 32, 256, 0, stream>>>(Obuf, mnb, ap, cv, d);
    fin_rstd<<<(d + 255) / 256, 256, 0, stream>>>(cv, rsb, d);

    apply_norm_pool<<<(NNODES * d) / 256, 256, 0, stream>>>(Obuf, mnb, rsb, gm, bt, ap,
                                                            Xbf, pool, d, pool_off);
    pool_off += d;
  }
  final_out<<<(NGRAPH * OUTDIM + 255) / 256, 256, 0, stream>>>(pool, (float*)d_out);
}
